// MultiHeadAttention_85572928405682
// MI455X (gfx1250) — matrix operand order repaired and checked by reading, not compiled, not run on hardware
//
#include <hip/hip_runtime.h>

#define HIDDEN   2048
#define HEADS    16
#define HEAD_DIM 128
#define BATCH    2
#define SEQ      2048
#define MTOT     (BATCH * SEQ)   // 4096 rows in every GEMM

typedef __bf16 bf16_t;
typedef __attribute__((ext_vector_type(4)))  __bf16 v4bf;
typedef __attribute__((ext_vector_type(8)))  __bf16 v8bf;
typedef __attribute__((ext_vector_type(16))) __bf16 v16bf;
typedef __attribute__((ext_vector_type(8)))  float  v8f;

// ---------- helpers ----------

__device__ inline bf16_t to_bf16(float f) {
  union { float f; unsigned u; } in; in.f = f;
  unsigned r = in.u + 0x7FFFu + ((in.u >> 16) & 1u);   // round-to-nearest-even
  union { unsigned short s; bf16_t b; } out;
  out.s = (unsigned short)(r >> 16);
  return out.b;
}

__device__ inline v8f zero8() {
  v8f z;
#pragma unroll
  for (int i = 0; i < 8; ++i) z[i] = 0.0f;
  return z;
}

// A-fragment (16x32 bf16): per lane 16 contiguous elems at K-offset
// koff (koff = (lane>>4)*16), row = lane&15.
__device__ inline v16bf load_a_frag(const bf16_t* p) {
  v8bf lo = *(const v8bf*)(p);
  v8bf hi = *(const v8bf*)(p + 8);
  v16bf r;
#pragma unroll
  for (int i = 0; i < 8; ++i) { r[i] = lo[i]; r[i + 8] = hi[i]; }
  return r;
}

__device__ inline v8f wmma_bf16(v16bf a, v16bf b, v8f c) {
  return __builtin_amdgcn_wmma_f32_16x16x32_bf16(
      /*neg_a=*/false, a, /*neg_b=*/false, b,
      /*c_mod=*/(short)0, c, /*reuse_a=*/false, /*reuse_b=*/false);
}

// ---------- fp32 -> bf16 cast (vectorized x4) ----------

__global__ void cvt_f32_to_bf16(const float* __restrict__ in,
                                bf16_t* __restrict__ out, int n4) {
  int i = blockIdx.x * blockDim.x + threadIdx.x;
  if (i < n4) {
    float4 v = ((const float4*)in)[i];
    v4bf o;
    o[0] = to_bf16(v.x); o[1] = to_bf16(v.y);
    o[2] = to_bf16(v.z); o[3] = to_bf16(v.w);
    ((v4bf*)out)[i] = o;
  }
}

// ---------- WMMA GEMM: Y = A(M x 2048 bf16 row-major) @ W^T (W is N x 2048 row-major) ----------
// One wave computes a 64x64 tile: 4 M-tiles x 4 N-tiles = 16 WMMAs per K-step of 32
// (~32 FLOP/byte of L2 traffic).
// mode 0: store bf16 to [B,H,S,Dh]   (Q, K)
// mode 1: store bf16 to [B,H,Dh,S]   (V transposed for the P*V stage)
// mode 2: store fp32 row-major [M, 2048] (final output projection)
__global__ __launch_bounds__(32) void gemm_bf16_wmma(
    const bf16_t* __restrict__ A, const bf16_t* __restrict__ W,
    bf16_t* __restrict__ outB, float* __restrict__ outF, int mode) {
  const int K = HIDDEN;
  int lane  = threadIdx.x;
  int m0    = blockIdx.x * 64;
  int n0    = blockIdx.y * 64;
  int arow  = lane & 15;
  int akoff = (lane >> 4) << 4;    // 0 or 16
  int bcol  = lane & 15;
  int bkoff = (lane >> 4) << 4;    // 0 or 16

  v8f acc[4][4];
#pragma unroll
  for (int mi = 0; mi < 4; ++mi)
#pragma unroll
    for (int t = 0; t < 4; ++t) acc[mi][t] = zero8();

  const bf16_t* ap = A + (size_t)(m0 + arow) * K + akoff;

  for (int k0 = 0; k0 < K; k0 += 32) {
    v16bf a[4];
#pragma unroll
    for (int mi = 0; mi < 4; ++mi)
      a[mi] = load_a_frag(ap + (size_t)(mi * 16) * K + k0);
    v16bf bf[4];
#pragma unroll
    for (int t = 0; t < 4; ++t)
      bf[t] = *(const v16bf*)(W + (size_t)(n0 + t * 16 + bcol) * K + k0 + bkoff);
#pragma unroll
    for (int mi = 0; mi < 4; ++mi)
#pragma unroll
      for (int t = 0; t < 4; ++t)
        acc[mi][t] = wmma_bf16(a[mi], bf[t], acc[mi][t]);
  }

  int rbase = (lane >> 4) << 3;    // 0 or 8 (C/D layout)
#pragma unroll
  for (int mi = 0; mi < 4; ++mi)
#pragma unroll
    for (int t = 0; t < 4; ++t)
#pragma unroll
      for (int r = 0; r < 8; ++r) {
        int row = m0 + mi * 16 + rbase + r;
        int col = n0 + t * 16 + (lane & 15);
        float v = acc[mi][t][r];
        if (mode == 2) {
          outF[(size_t)row * HIDDEN + col] = v;
        } else {
          int b = row >> 11, s = row & (SEQ - 1);
          int h = col >> 7,  d = col & (HEAD_DIM - 1);
          size_t idx;
          if (mode == 0)
            idx = (((size_t)(b * HEADS + h) * SEQ) + s) * HEAD_DIM + d;
          else
            idx = (((size_t)(b * HEADS + h) * HEAD_DIM) + d) * SEQ + s;
          outB[idx] = to_bf16(v);
        }
      }
}

// ---------- flash attention core: 1 wave per (b, h, 32-query tile) ----------
// Two 16-row query groups share every K/V block -> halves L2 K/V traffic.
// Q,K: [B,H,S,128] bf16;  Vt: [B,H,128,S] bf16;  Ctx: [B,S,2048] bf16
__global__ __launch_bounds__(32) void attn_flash_wmma(
    const bf16_t* __restrict__ Q, const bf16_t* __restrict__ Km,
    const bf16_t* __restrict__ Vt, bf16_t* __restrict__ Ctx) {
  int lane = threadIdx.x;
  int qt = blockIdx.x, h = blockIdx.y, b = blockIdx.z;
  int q0 = qt * 32;

  const bf16_t* q  = Q  + (size_t)(b * HEADS + h) * SEQ * HEAD_DIM;
  const bf16_t* kk = Km + (size_t)(b * HEADS + h) * SEQ * HEAD_DIM;
  const bf16_t* vt = Vt + (size_t)(b * HEADS + h) * HEAD_DIM * SEQ;

  int arow  = lane & 15;
  int akoff = (lane >> 4) << 4;
  int bcol  = lane & 15;
  int bkoff = (lane >> 4) << 4;
  int rbase = (lane >> 4) << 3;

  // Q tile: 2 groups x 4 A-fragments over d (128 = 4 * 32)
  v16bf qa[2][4];
#pragma unroll
  for (int g = 0; g < 2; ++g)
#pragma unroll
    for (int dt = 0; dt < 4; ++dt)
      qa[g][dt] =
          load_a_frag(q + (size_t)(q0 + g * 16 + arow) * HEAD_DIM + dt * 32 + akoff);

  v8f o[2][8];            // per group: 16x128 output accumulator
#pragma unroll
  for (int g = 0; g < 2; ++g)
#pragma unroll
    for (int t = 0; t < 8; ++t) o[g][t] = zero8();
  float mrow[2][8], lrow[2][8];
#pragma unroll
  for (int g = 0; g < 2; ++g)
#pragma unroll
    for (int r = 0; r < 8; ++r) { mrow[g][r] = -3.0e38f; lrow[g][r] = 0.0f; }

  __shared__ bf16_t lds_p[16][32];   // P tile bounce buffer (D-layout -> A-layout)

  const float scale = 0.08838834764831845f;  // 1/sqrt(128)
  int nkb = (q0 + 63) >> 5;                  // 32-key blocks covering keys <= q0+31

  for (int kb = 0; kb < nkb; ++kb) {
    int kbase = kb * 32;
    // prefetch next key block's K rows (one row head per lane)
    if (kb + 1 < nkb)
      __builtin_prefetch(kk + (size_t)(kbase + 32 + lane) * HEAD_DIM, 0, 1);

#pragma unroll
    for (int g = 0; g < 2; ++g) {
      // S = Qg @ K^T for this 32-key block (K frags reload per group; 2nd hits L0)
      v8f s0 = zero8(), s1 = zero8();
#pragma unroll
      for (int dt = 0; dt < 4; ++dt) {
        v16bf kf0 =
            *(const v16bf*)(kk + (size_t)(kbase + bcol) * HEAD_DIM + dt * 32 + bkoff);
        v16bf kf1 = *(const v16bf*)(kk + (size_t)(kbase + 16 + bcol) * HEAD_DIM +
                                    dt * 32 + bkoff);
        s0 = wmma_bf16(qa[g][dt], kf0, s0);
        s1 = wmma_bf16(qa[g][dt], kf1, s1);
      }

      // online softmax for this group's 16 rows
#pragma unroll
      for (int r = 0; r < 8; ++r) {
        int qm  = q0 + g * 16 + rbase + r;
        int kn0 = kbase + bcol;
        int kn1 = kn0 + 16;
        float v0 = (kn0 > qm) ? -3.0e38f : s0[r] * scale;
        float v1 = (kn1 > qm) ? -3.0e38f : s1[r] * scale;
        float bmax = fmaxf(v0, v1);
#pragma unroll
        for (int off = 8; off >= 1; off >>= 1)
          bmax = fmaxf(bmax, __shfl_xor(bmax, off, 32));
        float mnew = fmaxf(mrow[g][r], bmax);
        float p0 = __expf(v0 - mnew);
        float p1 = __expf(v1 - mnew);
        float ls = p0 + p1;
#pragma unroll
        for (int off = 8; off >= 1; off >>= 1)
          ls += __shfl_xor(ls, off, 32);
        float alpha = __expf(mrow[g][r] - mnew);
        lrow[g][r] = lrow[g][r] * alpha + ls;
        mrow[g][r] = mnew;
#pragma unroll
        for (int t = 0; t < 8; ++t) o[g][t][r] *= alpha;
        lds_p[rbase + r][bcol]      = to_bf16(p0);
        lds_p[rbase + r][bcol + 16] = to_bf16(p1);
      }
      __syncthreads();   // single-wave WG: cheap; orders LDS write -> read

      // P (16x32) as A-fragment; V block via transposed-V contiguous B-fragments
      v16bf pa = load_a_frag(&lds_p[arow][akoff]);
#pragma unroll
      for (int t = 0; t < 8; ++t) {
        v16bf bv = *(const v16bf*)(vt + (size_t)(t * 16 + bcol) * SEQ + kbase + bkoff);
        o[g][t] = wmma_bf16(pa, bv, o[g][t]);
      }
      __syncthreads();   // protect lds_p before next group's writes
    }
  }

  // epilogue: normalize and write context in [B,S,H*Dh] bf16
#pragma unroll
  for (int g = 0; g < 2; ++g)
#pragma unroll
    for (int r = 0; r < 8; ++r) {
      float inv = 1.0f / lrow[g][r];
      int row = q0 + g * 16 + rbase + r;
#pragma unroll
      for (int t = 0; t < 8; ++t) {
        int d = t * 16 + bcol;
        Ctx[((size_t)(b * SEQ + row)) * HIDDEN + h * HEAD_DIM + d] =
            to_bf16(o[g][t][r] * inv);
      }
    }
}

// ---------- host-side launch ----------

extern "C" void kernel_launch(void* const* d_in, const int* in_sizes, int n_in,
                              void* d_out, int out_size, void* d_ws, size_t ws_size,
                              hipStream_t stream) {
  const float* x  = (const float*)d_in[0];
  const float* wq = (const float*)d_in[1];
  const float* wk = (const float*)d_in[2];
  const float* wv = (const float*)d_in[3];
  const float* wo = (const float*)d_in[4];
  float* out = (float*)d_out;

  char* ws = (char*)d_ws;
  auto take = [&](size_t bytes) {
    char* p = ws;
    ws += (bytes + 255) & ~(size_t)255;
    return p;
  };
  const size_t xbytes = (size_t)MTOT * HIDDEN * sizeof(bf16_t);
  const size_t wbytes = (size_t)HIDDEN * HIDDEN * sizeof(bf16_t);
  const size_t hbytes = (size_t)BATCH * HEADS * SEQ * HEAD_DIM * sizeof(bf16_t);

  bf16_t* xb   = (bf16_t*)take(xbytes);
  bf16_t* wqb  = (bf16_t*)take(wbytes);
  bf16_t* wkb  = (bf16_t*)take(wbytes);
  bf16_t* wvb  = (bf16_t*)take(wbytes);
  bf16_t* wob  = (bf16_t*)take(wbytes);
  bf16_t* qb   = (bf16_t*)take(hbytes);   // [B,H,S,Dh]
  bf16_t* kbuf = (bf16_t*)take(hbytes);   // [B,H,S,Dh]
  bf16_t* vtb  = (bf16_t*)take(hbytes);   // [B,H,Dh,S]
  bf16_t* ctx  = (bf16_t*)take(xbytes);   // [B,S,H*Dh]

  int nx4 = (MTOT * HIDDEN) / 4;
  int nw4 = (HIDDEN * HIDDEN) / 4;
  cvt_f32_to_bf16<<<(nx4 + 255) / 256, 256, 0, stream>>>(x,  xb,  nx4);
  cvt_f32_to_bf16<<<(nw4 + 255) / 256, 256, 0, stream>>>(wq, wqb, nw4);
  cvt_f32_to_bf16<<<(nw4 + 255) / 256, 256, 0, stream>>>(wk, wkb, nw4);
  cvt_f32_to_bf16<<<(nw4 + 255) / 256, 256, 0, stream>>>(wv, wvb, nw4);
  cvt_f32_to_bf16<<<(nw4 + 255) / 256, 256, 0, stream>>>(wo, wob, nw4);

  dim3 gProj(MTOT / 64, HIDDEN / 64);
  gemm_bf16_wmma<<<gProj, 32, 0, stream>>>(xb, wqb, qb,   nullptr, 0);
  gemm_bf16_wmma<<<gProj, 32, 0, stream>>>(xb, wkb, kbuf, nullptr, 0);
  gemm_bf16_wmma<<<gProj, 32, 0, stream>>>(xb, wvb, vtb,  nullptr, 1);

  attn_flash_wmma<<<dim3(SEQ / 32, HEADS, BATCH), 32, 0, stream>>>(qb, kbuf, vtb, ctx);

  gemm_bf16_wmma<<<gProj, 32, 0, stream>>>(ctx, wob, nullptr, out, 2);
}
